// RNN_DRR_90924457656857
// MI455X (gfx1250) — compile-verified, matrix operand order checked by reading, not run on hardware
//
#include <hip/hip_runtime.h>

// ---------------- sizes ----------------
// B=256, T=512, IN=129, H=258, 3H=774
// padded: IN 129->160, H 258->288, 3H 774->784 (49 n-tiles), tower-out 129->144
// ws layout (in __bf16 elements):
#define WIH0_OFF 0u          // 784x160
#define WIH1_OFF 125440u     // 784x288
#define WIH2_OFF 351232u
#define WHH0_OFF 577024u
#define WHH1_OFF 802816u
#define WHH2_OFF 1028608u
#define T1W1_OFF 1254400u    // 144x288
#define T1W2_OFF 1295872u    // 144x160
#define T2W1_OFF 1318912u
#define T2W2_OFF 1360384u
#define OUT0_OFF 1383424u    // 131072x288 bf16 (recurrent output, tower input)

#define ROUT2_OFF 16908288u  // d_out float offsets
#define HNEW_OFF  33816576u

typedef __attribute__((ext_vector_type(16))) __bf16 bf16x16;
typedef __attribute__((ext_vector_type(8)))  __bf16 bf16x8;
typedef __attribute__((ext_vector_type(8)))  float  f32x8;
typedef __attribute__((ext_vector_type(4)))  unsigned int u32x4;
typedef __attribute__((ext_vector_type(8)))  int i32x8;
typedef __attribute__((ext_vector_type(4)))  int i32x4;
typedef unsigned int u32;
typedef unsigned long long u64;

// explicit address-space pointer types: force global_load/ds_load (never flat)
typedef const __attribute__((address_space(1))) __bf16*  gcbf;
typedef const __attribute__((address_space(3))) __bf16*  lcbf;
typedef const __attribute__((address_space(1))) bf16x8*  gcbf8;
typedef const __attribute__((address_space(3))) bf16x8*  lcbf8;
typedef const __attribute__((address_space(1))) float*   gcf;
typedef __attribute__((address_space(1))) float*         gf;
typedef __attribute__((address_space(1))) __bf16*        gbf;

union FragBF { bf16x16 v; bf16x8 h[2]; };

__device__ __forceinline__ __bf16 f2bf(float f) {
  union { float f; unsigned u; } a; a.f = f;
  unsigned r = (a.u + 0x7FFFu + ((a.u >> 16) & 1u)) >> 16;
  union { unsigned short s; __bf16 b; } o; o.s = (unsigned short)r;
  return o.b;
}
__device__ __forceinline__ float bf2f(__bf16 b) {
  union { unsigned short s; __bf16 b; } i; i.b = b;
  union { float f; unsigned u; } o; o.u = ((unsigned)i.s) << 16;
  return o.f;
}
__device__ __forceinline__ float sigf(float x) { return 1.0f / (1.0f + __expf(-x)); }

__device__ __forceinline__ void ldfragG(FragBF& f, gcbf p, int kb, int hi) {
  f.h[0] = *(gcbf8)(p + kb + hi * 8);
  f.h[1] = *(gcbf8)(p + kb + 16 + hi * 8);
}
__device__ __forceinline__ void ldfragL(FragBF& f, lcbf p, int kb, int hi) {
  f.h[0] = *(lcbf8)(p + kb + hi * 8);
  f.h[1] = *(lcbf8)(p + kb + 16 + hi * 8);
}

// 2-stage software-pipelined K loop: A from LDS, B from global.
template<int KC, bool PF>
__device__ __forceinline__ f32x8 gemm_kloop(lcbf Ar, gcbf Br, int hi) {
  FragBF a0, b0, a1, b1;
  f32x8 acc = {};
  ldfragL(a0, Ar, 0, hi);
  ldfragG(b0, Br, 0, hi);
  #pragma unroll
  for (int k = 0; k < KC; ++k) {
    if (k + 1 < KC) {
      ldfragL(a1, Ar, (k + 1) * 32, hi);
      ldfragG(b1, Br, (k + 1) * 32, hi);
      if (PF) __builtin_prefetch((const void*)(Br + (k + 2) * 32), 0, 3);
    }
    acc = __builtin_amdgcn_wmma_f32_16x16x32_bf16(
        false, a0.v, false, b0.v, (short)0, acc, false, false);
    if (k + 1 < KC) { a0 = a1; b0 = b1; }
  }
  return acc;
}

// ---------------- TDM: 2D tile global -> LDS ----------------
#if __has_builtin(__builtin_amdgcn_tensor_load_to_lds)
#define HAVE_TDM 1
__device__ __forceinline__ void tdm_load_2d(u32 lds_byte_off, const void* gptr,
                                            u32 tensor_d0, u32 tensor_d1,
                                            u32 tile_d0, u32 tile_d1, u64 d0_stride) {
  u64 ga = (u64)(size_t)gptr;
  u32x4 g0;
  g0[0] = 1u;                                   // count=1 (valid), user mode, no gather
  g0[1] = lds_byte_off;                         // lds_addr (bytes)
  g0[2] = (u32)ga;                              // global_addr[31:0]
  g0[3] = ((u32)(ga >> 32) & 0x01FFFFFFu) | (2u << 30);  // addr[56:32] | type=2
  i32x8 g1;
  g1[0] = (int)(1u << 16);                      // data_size=1 (2 bytes, bf16)
  g1[1] = (int)((tensor_d0 & 0xFFFFu) << 16);   // tensor_dim0[15:0]
  g1[2] = (int)(((tensor_d0 >> 16) & 0xFFFFu) | ((tensor_d1 & 0xFFFFu) << 16));
  g1[3] = (int)(((tensor_d1 >> 16) & 0xFFFFu) | ((tile_d0 & 0xFFFFu) << 16));
  g1[4] = (int)(tile_d1 & 0xFFFFu);             // tile_dim1 | tile_dim2=0
  g1[5] = (int)(u32)(d0_stride & 0xFFFFFFFFu);  // tensor_dim0_stride[31:0]
  g1[6] = (int)(u32)((d0_stride >> 32) & 0xFFFFu);  // stride[47:32] | dim1_stride lo=0
  g1[7] = 0;
  i32x4 g2 = {0, 0, 0, 0};                      // 2D: dims 2/3 unused
  i32x4 g3 = {0, 0, 0, 0};
#if defined(__clang_major__) && (__clang_major__ >= 23)
  i32x8 g4 = {0, 0, 0, 0, 0, 0, 0, 0};
  __builtin_amdgcn_tensor_load_to_lds(g0, g1, g2, g3, g4, 0);
#else
  __builtin_amdgcn_tensor_load_to_lds(g0, g1, g2, g3, 0);
#endif
}
#else
#define HAVE_TDM 0
#endif

// ---------------- weight f32 -> padded bf16 ----------------
__global__ void cvt_pad(const float* __restrict__ src, __bf16* __restrict__ dst,
                        int rows, int cols, int rpad, int cpad) {
  gcf s = (gcf)src;
  gbf d = (gbf)dst;
  int n = rpad * cpad;
  for (int i = blockIdx.x * blockDim.x + threadIdx.x; i < n; i += gridDim.x * blockDim.x) {
    int r = i / cpad, c = i % cpad;
    float v = (r < rows && c < cols) ? s[r * cols + c] : 0.0f;
    d[i] = f2bf(v);
  }
}

// ---------------- persistent GRU: 16 WGs, one per 16-row batch tile ----------------
__global__ __launch_bounds__(1024) void gru_persistent(
    const float* __restrict__ x, const float* __restrict__ h_state,
    const float* __restrict__ bih0, const float* __restrict__ bhh0,
    const float* __restrict__ bih1, const float* __restrict__ bhh1,
    const float* __restrict__ bih2, const float* __restrict__ bhh2,
    __bf16* __restrict__ ws, float* __restrict__ dout) {
  extern __shared__ char smem[];
  __bf16* h_bf = (__bf16*)smem;            // [3][16][288] bf16
  __bf16* xin  = h_bf + 3 * 16 * 288;      // [16][160]    bf16
  float*  gxb  = (float*)(xin + 16 * 160); // [16][784]    f32
  float*  ghb  = gxb + 16 * 784;           // [16][784]    f32
  float*  bI   = ghb + 16 * 784;           // [3][774]     f32 (LDS-staged biases)
  float*  bH   = bI + 3 * 774;             // [3][774]     f32

  const int tid  = threadIdx.x;
  const int wave = tid >> 5;
  const int lane = tid & 31;
  const int m    = lane & 15;
  const int hi   = lane >> 4;
  const int b0   = blockIdx.x * 16;

  lcbf h_bfL = (lcbf)h_bf;
  lcbf xinL  = (lcbf)xin;
  const gcbf wihG[3] = { (gcbf)(ws + WIH0_OFF), (gcbf)(ws + WIH1_OFF), (gcbf)(ws + WIH2_OFF) };
  const gcbf whhG[3] = { (gcbf)(ws + WHH0_OFF), (gcbf)(ws + WHH1_OFF), (gcbf)(ws + WHH2_OFF) };
  const gcf bihG[3] = { (gcf)bih0, (gcf)bih1, (gcf)bih2 };
  const gcf bhhG[3] = { (gcf)bhh0, (gcf)bhh1, (gcf)bhh2 };
  gcf xg  = (gcf)x;
  gcf hsg = (gcf)h_state;
  gbf out0   = (gbf)(ws + OUT0_OFF);
  gf  dout_h = (gf)(dout + HNEW_OFF);

  // init hidden state (bf16, zero-padded cols 258..287) + stage biases into LDS
  for (int i = tid; i < 3 * 16 * 288; i += 1024) {
    int l = i / (16 * 288);
    int r = (i / 288) % 16;
    int c = i % 288;
    float v = (c < 258) ? hsg[((size_t)l * 256 + b0 + r) * 258 + c] : 0.0f;
    h_bf[i] = f2bf(v);
  }
  for (int i = tid; i < 3 * 774; i += 1024) {
    bI[i] = bihG[i / 774][i % 774];
    bH[i] = bhhG[i / 774][i % 774];
  }
  __syncthreads();

  for (int t = 0; t < 512; ++t) {
    // stage x_t for our 16 rows (bf16, zero-padded to 160)
    for (int i = tid; i < 16 * 160; i += 1024) {
      int r = i / 160, c = i % 160;
      float v = (c < 129) ? xg[((size_t)(b0 + r) * 512 + t) * 129 + c] : 0.0f;
      xin[i] = f2bf(v);
    }
    __syncthreads();

    for (int l = 0; l < 3; ++l) {
      // 98 tile jobs: 0..48 -> gx = in . Wih^T ; 49..97 -> gh = h . Whh^T
      for (int jj = wave; jj < 98; jj += 32) {
        const bool is_gh = (jj >= 49);
        const int nt = is_gh ? (jj - 49) : jj;
        lcbf Abase; int sA; int kc; gcbf Wbase;
        if (is_gh)       { Abase = h_bfL + l * 16 * 288;       sA = 288; kc = 9; Wbase = whhG[l]; }
        else if (l == 0) { Abase = xinL;                       sA = 160; kc = 5; Wbase = wihG[0]; }
        else             { Abase = h_bfL + (l - 1) * 16 * 288; sA = 288; kc = 9; Wbase = wihG[l]; }
        lcbf Ar = Abase + m * sA;
        gcbf Br = Wbase + (size_t)(nt * 16 + m) * sA;
        f32x8 acc;
        if (kc == 9) acc = gemm_kloop<9, true>(Ar, Br, hi);
        else         acc = gemm_kloop<5, true>(Ar, Br, hi);
        float* G = is_gh ? ghb : gxb;
        const int cb = nt * 16 + m;
        #pragma unroll
        for (int r = 0; r < 8; ++r) G[(r + 8 * hi) * 784 + cb] = acc[r];
      }
      __syncthreads();

      // elementwise GRU update: r,z,n gates; biases from LDS
      const float* bi = bI + l * 774;
      const float* bh = bH + l * 774;
      for (int idx = tid; idx < 16 * 288; idx += 1024) {
        int b = idx / 288, c = idx % 288;
        if (c < 258) {
          float rg = sigf(gxb[b * 784 + c]       + bi[c]       + ghb[b * 784 + c]       + bh[c]);
          float zg = sigf(gxb[b * 784 + c + 258] + bi[c + 258] + ghb[b * 784 + c + 258] + bh[c + 258]);
          float hn = ghb[b * 784 + c + 516] + bh[c + 516];
          float ng = tanhf(gxb[b * 784 + c + 516] + bi[c + 516] + rg * hn);
          float hp = bf2f(h_bf[(l * 16 + b) * 288 + c]);
          float hv = (1.0f - zg) * ng + zg * hp;
          h_bf[(l * 16 + b) * 288 + c] = f2bf(hv);
          if (l == 2) out0[((size_t)(b0 + b) * 512 + t) * 288 + c] = f2bf(hv);
          if (t == 511) dout_h[((size_t)l * 256 + (b0 + b)) * 258 + c] = hv;
        } else if (l == 2) {
          out0[((size_t)(b0 + b) * 512 + t) * 288 + c] = f2bf(0.0f);
        }
      }
      __syncthreads();
    }
  }
}

// ---------------- towers: fused 2-layer MLP over M = B*T = 131072 rows ----------------
// LDS: A-tile [128][288] bf16 (TDM-staged, 73728 B) + h1 [128][160] bf16 (40960 B)
__global__ __launch_bounds__(256) void tower_kernel(
    const __bf16* __restrict__ ws,
    const float* __restrict__ t1b1, const float* __restrict__ t1b2,
    const float* __restrict__ t2b1, const float* __restrict__ t2b2,
    float* __restrict__ dout) {
  extern __shared__ char tsm[];
  __bf16* aT = (__bf16*)tsm;        // [128][288]
  __bf16* h1 = aT + 128 * 288;      // [128][160]
  lcbf aTL = (lcbf)aT;
  lcbf h1L = (lcbf)h1;

  const int tid  = threadIdx.x;
  const int wave = tid >> 5;
  const int lane = tid & 31;
  const int m    = lane & 15;
  const int hi   = lane >> 4;
  const size_t m0 = (size_t)blockIdx.x * 128;
  const __bf16* out0 = ws + OUT0_OFF;

  // Stage the 128x288 A-tile once; it feeds both towers' stage-1 GEMMs.
#if HAVE_TDM
  if (tid < 32) {  // one wave issues the DMA (TDM ignores EXEC within the wave)
    tdm_load_2d((u32)(size_t)(void*)aT, out0 + m0 * 288,
                /*tensor_d0=*/288, /*tensor_d1=*/128,
                /*tile_d0=*/288, /*tile_d1=*/128, /*d0_stride=*/288);
    __builtin_amdgcn_s_wait_tensorcnt(0);
  }
#else
  {
    gcbf src = (gcbf)(out0 + m0 * 288);
    for (int i = tid; i < 128 * 288; i += 256) aT[i] = src[i];
  }
#endif
  for (int i = tid; i < 128 * 160; i += 256) h1[i] = f2bf(0.0f);
  __syncthreads();

  const gcbf W1s[2] = { (gcbf)(ws + T1W1_OFF), (gcbf)(ws + T2W1_OFF) };
  const gcbf W2s[2] = { (gcbf)(ws + T1W2_OFF), (gcbf)(ws + T2W2_OFF) };
  const gcf  B1s[2] = { (gcf)t1b1, (gcf)t2b1 };
  const gcf  B2s[2] = { (gcf)t1b2, (gcf)t2b2 };
  gf dsts[2] = { (gf)dout, (gf)(dout + ROUT2_OFF) };

  for (int tower = 0; tower < 2; ++tower) {
    // stage 1: h1 = relu(aT[128,288] . W1^T[288,144] + b1) -> LDS bf16
    const gcbf W1 = W1s[tower]; const gcf b1 = B1s[tower];
    for (int job = wave; job < 72; job += 8) {
      int mt = job / 9, nt = job % 9;
      lcbf Ar = aTL + (mt * 16 + m) * 288;
      gcbf Br = W1 + (size_t)(nt * 16 + m) * 288;
      f32x8 acc = gemm_kloop<9, true>(Ar, Br, hi);
      int n = nt * 16 + m;
      float bias = (n < 129) ? b1[n] : 0.0f;
      #pragma unroll
      for (int r = 0; r < 8; ++r) {
        float v = fmaxf(acc[r] + bias, 0.0f);
        h1[(mt * 16 + r + 8 * hi) * 160 + n] = f2bf(v);
      }
    }
    __syncthreads();

    // stage 2: out = act(h1[128,160] . W2^T[160,144] + b2) -> d_out
    const gcbf W2 = W2s[tower]; const gcf b2 = B2s[tower];
    gf dst = dsts[tower];
    for (int job = wave; job < 72; job += 8) {
      int mt = job / 9, nt = job % 9;
      lcbf Ar = h1L + (mt * 16 + m) * 160;
      gcbf Br = W2 + (size_t)(nt * 16 + m) * 160;
      f32x8 acc = gemm_kloop<5, true>(Ar, Br, hi);
      int n = nt * 16 + m;
      if (n < 129) {
        float bias = b2[n];
        #pragma unroll
        for (int r = 0; r < 8; ++r) {
          float v = acc[r] + bias;
          if (tower == 0) v = sigf(v);
          dst[(m0 + (size_t)mt * 16 + r + 8 * hi) * 129 + n] = v;
        }
      }
    }
    __syncthreads();  // h1 re-written by next tower's stage 1
  }
}

extern "C" void kernel_launch(void* const* d_in, const int* in_sizes, int n_in,
                              void* d_out, int out_size, void* d_ws, size_t ws_size,
                              hipStream_t stream) {
  (void)in_sizes; (void)n_in; (void)out_size; (void)ws_size;
  const float* x  = (const float*)d_in[0];
  const float* hs = (const float*)d_in[1];
  const float* wih_[3] = { (const float*)d_in[2], (const float*)d_in[6],  (const float*)d_in[10] };
  const float* whh_[3] = { (const float*)d_in[3], (const float*)d_in[7],  (const float*)d_in[11] };
  const float* bih_[3] = { (const float*)d_in[4], (const float*)d_in[8],  (const float*)d_in[12] };
  const float* bhh_[3] = { (const float*)d_in[5], (const float*)d_in[9],  (const float*)d_in[13] };
  const float* t1w1 = (const float*)d_in[14]; const float* t1b1 = (const float*)d_in[15];
  const float* t1w2 = (const float*)d_in[16]; const float* t1b2 = (const float*)d_in[17];
  const float* t2w1 = (const float*)d_in[18]; const float* t2b1 = (const float*)d_in[19];
  const float* t2w2 = (const float*)d_in[20]; const float* t2b2 = (const float*)d_in[21];
  __bf16* ws = (__bf16*)d_ws;
  float* out = (float*)d_out;

  // Phase 0: weight conversion to padded bf16
  struct CvtJob { const float* src; unsigned off; int r, c, rp, cp; };
  const CvtJob jobs[10] = {
    { wih_[0], WIH0_OFF, 774, 129, 784, 160 },
    { wih_[1], WIH1_OFF, 774, 258, 784, 288 },
    { wih_[2], WIH2_OFF, 774, 258, 784, 288 },
    { whh_[0], WHH0_OFF, 774, 258, 784, 288 },
    { whh_[1], WHH1_OFF, 774, 258, 784, 288 },
    { whh_[2], WHH2_OFF, 774, 258, 784, 288 },
    { t1w1,    T1W1_OFF, 129, 258, 144, 288 },
    { t1w2,    T1W2_OFF, 129, 129, 144, 160 },
    { t2w1,    T2W1_OFF, 129, 258, 144, 288 },
    { t2w2,    T2W2_OFF, 129, 129, 144, 160 },
  };
  for (int i = 0; i < 10; ++i) {
    int nel = jobs[i].rp * jobs[i].cp;
    cvt_pad<<<(nel + 255) / 256, 256, 0, stream>>>(jobs[i].src, ws + jobs[i].off,
                                                   jobs[i].r, jobs[i].c, jobs[i].rp, jobs[i].cp);
  }

  // Phase 1: persistent recurrence (16 independent batch tiles, no global sync)
  size_t smem = (size_t)(3 * 16 * 288 + 16 * 160) * sizeof(__bf16)
              + (size_t)(2 * 16 * 784 + 2 * 3 * 774) * sizeof(float); // 151696 B (< 320KB/WGP)
  gru_persistent<<<16, 1024, smem, stream>>>(x, hs,
      bih_[0], bhh_[0], bih_[1], bhh_[1], bih_[2], bhh_[2], ws, out);

  // Phase 2: towers over all B*T rows, full-chip GEMM
  size_t tsmem = (size_t)(128 * 288 + 128 * 160) * sizeof(__bf16); // 114688 B
  tower_kernel<<<1024, 256, tsmem, stream>>>(ws, t1b1, t1b2, t2b1, t2b2, out);
}
